// SAGE_23931557773766
// MI455X (gfx1250) — matrix operand order repaired and checked by reading, not compiled, not run on hardware
//
#include <hip/hip_runtime.h>
#include <hip/hip_bf16.h>
#include <stdint.h>

typedef __attribute__((ext_vector_type(16))) _Float16 v16h;
typedef __attribute__((ext_vector_type(8)))  float    v8f;

#define NODES_PER_TILE 16
#define K_TOT 256   // [aggr(128) | x(128)]
#define D_IN 128
#define A_STRIDE 260   // f32 row stride in LDS: 1040B (16B aligned, 4-bank skew mod 64)

// ---------------------------------------------------------------------------
// Degree accumulation: one thread per edge, f32 atomic into deg[dst].
// ---------------------------------------------------------------------------
__global__ void degree_kernel(const int* __restrict__ dst, float* __restrict__ deg, int E) {
    int e = blockIdx.x * blockDim.x + threadIdx.x;
    if (e < E) atomicAdd(&deg[dst[e]], 1.0f);
}

__global__ void invdeg_kernel(float* __restrict__ deg, int n) {
    int i = blockIdx.x * blockDim.x + threadIdx.x;
    if (i < n) deg[i] = 1.0f / fmaxf(deg[i], 1.0f);
}

// ---------------------------------------------------------------------------
// Scatter-add: one wave (32 lanes) per edge; lane owns 4 contiguous floats.
// Row = 512B -> fully coalesced float4 read of x[src], 4 atomic f32 adds to
// summ[dst] (L2-resident RMW on MI455X's 192MB L2).
// ---------------------------------------------------------------------------
__global__ void scatter_add128(const float* __restrict__ x, const int* __restrict__ src,
                               const int* __restrict__ dst, float* __restrict__ summ, int E) {
    const int lane = threadIdx.x & 31;
    const int wave = threadIdx.x >> 5;
    const int e = blockIdx.x * 8 + wave;
    if (e >= E) return;
    const int s = src[e];
    const int d = dst[e];
    const float4 v = ((const float4*)(x + (size_t)s * D_IN))[lane];
    float* o = summ + (size_t)d * D_IN + lane * 4;
    atomicAdd(o + 0, v.x);
    atomicAdd(o + 1, v.y);
    atomicAdd(o + 2, v.z);
    atomicAdd(o + 3, v.w);
}

// ---------------------------------------------------------------------------
// Pack combined weights [Wl; Wr] (K=256, d_out cols) into per-lane WMMA
// B-fragment order: element i of lane (n = lane&15, h = lane>>4) for k-tile kt,
// col-tile ct is  B[32*kt + 8*h + (i&7) + 16*(i>>3)][16*ct + n]  (ISA 7.12.2, 16-bit B 32x16).
// Linear layout: (((kt*CT + ct)*32 + lane)*16 + i) -> lane's 16 halves are
// contiguous => fragment load is two global_load_b128.
// ---------------------------------------------------------------------------
__global__ void prep_bfrag(const float* __restrict__ Wl, const float* __restrict__ Wr,
                           _Float16* __restrict__ bfrag, int CT) {
    const int total = 8 * CT * 512;
    int t = blockIdx.x * blockDim.x + threadIdx.x;
    if (t >= total) return;
    const int i    = t & 15;
    const int lane = (t >> 4) & 31;
    const int ct   = (t >> 9) % CT;
    const int kt   = t / (CT * 512);
    const int n = lane & 15, h = lane >> 4;
    const int k = 32 * kt + 8 * h + (i & 7) + ((i >> 3) << 4);
    const int o = ct * 16 + n;
    const float v = (k < D_IN) ? Wl[o * D_IN + k] : Wr[o * D_IN + (k - D_IN)];
    bfrag[t] = (_Float16)v;
}

// ---------------------------------------------------------------------------
// Fused SAGE layer GEMM:  out[16 x (CT*16)] = [aggr | x] (16 x 256) @ Bfrag
// Block = 128 threads (4 waves). Wave w computes col tiles {w (+4 if CT==8)}.
// A tile staged raw-f32 into LDS via GLOBAL_LOAD_ASYNC_TO_LDS_B128 (ASYNCcnt),
// then per k-tile: 4x ds_load_b128 + invdeg scale (aggr half only, compile
// time) + cvt to f16 fragment. 8 x v_wmma_f32_16x16x32_f16 per col tile.
// ---------------------------------------------------------------------------
template<int CT, bool RELU>
__global__ __launch_bounds__(128) void sage_gemm(
        const float* __restrict__ summ, const float* __restrict__ invdeg,
        const float* __restrict__ xin,  const _Float16* __restrict__ bfrag,
        const float* __restrict__ bias, float* __restrict__ out) {
    __shared__ float a32[NODES_PER_TILE][A_STRIDE];

    const int tid = threadIdx.x;
    const int node_base = blockIdx.x * NODES_PER_TILE;

    // ---- async stage: 16 rows x 256 f32 = 1024 float4 chunks, 8 rounds ----
    const uint32_t lds_base = (uint32_t)(uintptr_t)&a32[0][0];
    #pragma unroll
    for (int it = 0; it < 8; ++it) {
        const int c  = it * 128 + tid;   // chunk id
        const int r  = c >> 6;           // row 0..15
        const int k  = (c & 63) << 2;    // float index in row, 0..252
        const uint32_t lds_off = lds_base + (uint32_t)(r * A_STRIDE + k) * 4u;
        // wave-uniform branch: a wave's 32 lanes span one 128-float half
        if (k < D_IN) {
            const uint32_t goff = ((uint32_t)(node_base + r) * D_IN + (uint32_t)k) * 4u;
            asm volatile("global_load_async_to_lds_b128 %0, %1, %2"
                         :: "v"(lds_off), "v"(goff), "s"(summ) : "memory");
        } else {
            const uint32_t goff = ((uint32_t)(node_base + r) * D_IN + (uint32_t)(k - D_IN)) * 4u;
            asm volatile("global_load_async_to_lds_b128 %0, %1, %2"
                         :: "v"(lds_off), "v"(goff), "s"(xin) : "memory");
        }
    }
    asm volatile("s_wait_asynccnt 0x0" ::: "memory");
    __syncthreads();

    const int wave = tid >> 5;
    const int lane = tid & 31;
    const int n = lane & 15;     // A: row M; B/C/D: column N
    const int h = lane >> 4;     // half-wave group

    const float idg = invdeg[node_base + n];   // scale for A rows (aggr half)

    constexpr int TPW = CT / 4;  // col tiles per wave (2 or 1)
    v8f acc[TPW];
    #pragma unroll
    for (int j = 0; j < TPW; ++j) acc[j] = (v8f){};

    #pragma unroll
    for (int kt = 0; kt < 8; ++kt) {
        const int k0 = kt * 32;
        // A fragment (16-bit A 16x32 layout): elems 0..7 -> K=k0+8h+i,
        // elems 8..15 -> K=k0+16+8h+i. Four aligned 16B LDS loads (f32).
        const float* ap = &a32[n][k0 + 8 * h];
        float4 f0 = *(const float4*)(ap + 0);
        float4 f1 = *(const float4*)(ap + 4);
        float4 f2 = *(const float4*)(ap + 16);
        float4 f3 = *(const float4*)(ap + 20);
        if (kt < 4) {   // compile-time: k-tiles 0..3 are the aggr half -> mean scale
            f0.x *= idg; f0.y *= idg; f0.z *= idg; f0.w *= idg;
            f1.x *= idg; f1.y *= idg; f1.z *= idg; f1.w *= idg;
            f2.x *= idg; f2.y *= idg; f2.z *= idg; f2.w *= idg;
            f3.x *= idg; f3.y *= idg; f3.z *= idg; f3.w *= idg;
        }
        v16h a;
        a[0]  = (_Float16)f0.x; a[1]  = (_Float16)f0.y;
        a[2]  = (_Float16)f0.z; a[3]  = (_Float16)f0.w;
        a[4]  = (_Float16)f1.x; a[5]  = (_Float16)f1.y;
        a[6]  = (_Float16)f1.z; a[7]  = (_Float16)f1.w;
        a[8]  = (_Float16)f2.x; a[9]  = (_Float16)f2.y;
        a[10] = (_Float16)f2.z; a[11] = (_Float16)f2.w;
        a[12] = (_Float16)f3.x; a[13] = (_Float16)f3.y;
        a[14] = (_Float16)f3.z; a[15] = (_Float16)f3.w;

        #pragma unroll
        for (int j = 0; j < TPW; ++j) {
            const int ct = wave + j * 4;
            const v16h b = *(const v16h*)(bfrag + (((size_t)(kt * CT + ct) * 32 + lane) * 16));
            acc[j] = __builtin_amdgcn_wmma_f32_16x16x32_f16(
                    false, a, false, b, (short)0, acc[j], false, false);
        }
    }

    // Epilogue: C/D layout -> VGPR r holds D[M = r + 8h][N = n].
    constexpr int dout = CT * 16;
    #pragma unroll
    for (int j = 0; j < TPW; ++j) {
        const int colbase = (wave + j * 4) * 16;
        const float bv = bias[colbase + n];
        #pragma unroll
        for (int r = 0; r < 8; ++r) {
            const int node = node_base + r + 8 * h;
            float v = acc[j][r] + bv;
            if (RELU) v = fmaxf(v, 0.0f);
            out[(size_t)node * dout + colbase + n] = v;
        }
    }
}

// ---------------------------------------------------------------------------
// In-place log_softmax over 64 columns; one wave per node, shuffle reductions.
// ---------------------------------------------------------------------------
__global__ void log_softmax64(float* __restrict__ hbuf, int n_nodes) {
    const int lane = threadIdx.x & 31;
    const int wave = threadIdx.x >> 5;
    const int node = blockIdx.x * 8 + wave;
    if (node >= n_nodes) return;
    float* row = hbuf + (size_t)node * 64;
    const float a = row[lane];
    const float b = row[lane + 32];
    float mx = fmaxf(a, b);
    #pragma unroll
    for (int off = 16; off; off >>= 1) mx = fmaxf(mx, __shfl_xor(mx, off, 32));
    float s = expf(a - mx) + expf(b - mx);
    #pragma unroll
    for (int off = 16; off; off >>= 1) s += __shfl_xor(s, off, 32);
    const float lse = mx + logf(s);
    row[lane]      = a - lse;
    row[lane + 32] = b - lse;
}

// ---------------------------------------------------------------------------
extern "C" void kernel_launch(void* const* d_in, const int* in_sizes, int n_in,
                              void* d_out, int out_size, void* d_ws, size_t ws_size,
                              hipStream_t stream) {
    const float* x    = (const float*)d_in[0];
    const int*   edge = (const int*)d_in[1];   // [2,E] flat: src=edge[0:E], dst=edge[E:2E]
    const float* Wl0  = (const float*)d_in[2];
    const float* bl0  = (const float*)d_in[3];
    const float* Wr0  = (const float*)d_in[4];
    const float* Wl1  = (const float*)d_in[5];
    const float* bl1  = (const float*)d_in[6];
    const float* Wr1  = (const float*)d_in[7];
    const float* Wl2  = (const float*)d_in[8];
    const float* bl2  = (const float*)d_in[9];
    const float* Wr2  = (const float*)d_in[10];

    const int N = in_sizes[0] / D_IN;   // 100000 (divisible by 16)
    const int E = in_sizes[1] / 2;      // 1600000
    float* out = (float*)d_out;         // N x 64

    // Workspace carve-up (256B aligned slices).
    char* ws = (char*)d_ws;
    size_t off = 0;
    auto carve = [&](size_t bytes) -> void* {
        void* p = ws + off;
        off += (bytes + 255) & ~(size_t)255;
        return p;
    };
    float*    deg  = (float*)carve((size_t)N * 4);
    float*    summ = (float*)carve((size_t)N * D_IN * 4);
    float*    h0   = (float*)carve((size_t)N * D_IN * 4);
    float*    h1   = (float*)carve((size_t)N * D_IN * 4);
    _Float16* bf0  = (_Float16*)carve((size_t)8 * 8 * 512 * 2);   // CT=8
    _Float16* bf1  = (_Float16*)carve((size_t)8 * 8 * 512 * 2);   // CT=8
    _Float16* bf2  = (_Float16*)carve((size_t)8 * 4 * 512 * 2);   // CT=4

    const int* src = edge;
    const int* dst = edge + E;

    // Degrees (shared by all layers) -> inverse in place.
    hipMemsetAsync(deg, 0, (size_t)N * 4, stream);
    degree_kernel<<<(E + 255) / 256, 256, 0, stream>>>(dst, deg, E);
    invdeg_kernel<<<(N + 255) / 256, 256, 0, stream>>>(deg, N);

    // Pack weights into WMMA B-fragment order (L2-resident, reused by all tiles).
    prep_bfrag<<<(8 * 8 * 512 + 255) / 256, 256, 0, stream>>>(Wl0, Wr0, bf0, 8);
    prep_bfrag<<<(8 * 8 * 512 + 255) / 256, 256, 0, stream>>>(Wl1, Wr1, bf1, 8);
    prep_bfrag<<<(8 * 4 * 512 + 255) / 256, 256, 0, stream>>>(Wl2, Wr2, bf2, 4);

    const int scatter_blocks = (E + 7) / 8;
    const int gemm_blocks    = N / NODES_PER_TILE;   // N % 16 == 0

    // Layer 0: aggr(x) -> h0 = relu([aggr|x] @ [Wl0;Wr0]^T + bl0)
    hipMemsetAsync(summ, 0, (size_t)N * D_IN * 4, stream);
    scatter_add128<<<scatter_blocks, 256, 0, stream>>>(x, src, dst, summ, E);
    sage_gemm<8, true><<<gemm_blocks, 128, 0, stream>>>(summ, deg, x, bf0, bl0, h0);

    // Layer 1
    hipMemsetAsync(summ, 0, (size_t)N * D_IN * 4, stream);
    scatter_add128<<<scatter_blocks, 256, 0, stream>>>(h0, src, dst, summ, E);
    sage_gemm<8, true><<<gemm_blocks, 128, 0, stream>>>(summ, deg, h0, bf1, bl1, h1);

    // Layer 2 (no relu) -> writes straight into d_out, then in-place log_softmax.
    hipMemsetAsync(summ, 0, (size_t)N * D_IN * 4, stream);
    scatter_add128<<<scatter_blocks, 256, 0, stream>>>(h1, src, dst, summ, E);
    sage_gemm<4, false><<<gemm_blocks, 128, 0, stream>>>(summ, deg, h1, bf2, bl2, out);

    log_softmax64<<<(N + 7) / 8, 256, 0, stream>>>(out, N);
}
